// MARNNBaseEncoder_24309514895398
// MI455X (gfx1250) — matrix operand order repaired and checked by reading, not compiled, not run on hardware
//
#include <hip/hip_runtime.h>
#include <cstdint>

// ---------------- problem constants ----------------
#define T_STEPS   1024
#define BATCH     128
#define IDIM      256
#define CDIM      512
#define NSLOT     128
#define MDIM      64
#define KTOT      832      // IDIM + MDIM + CDIM
#define KIH       320      // IDIM + MDIM
#define NGROUP    8        // batch groups of 16
#define NCW       16       // column-workgroups per group
#define GCOLS     128      // gate columns per CW (32 hidden units x 4 gates)
#define HSLICE    32
#define KP        840      // padded K stride (bf16 elements) for W/X in LDS (conflict-free)
#define HP        40       // padded stride for 32-wide bf16 LDS rows

// ---------------- LDS layout (bytes), all 16-aligned ----------------
#define W_OFF     0
#define W_BYTES   (GCOLS*KP*2)            // 215040
#define X_OFF     (W_OFF + W_BYTES)
#define X_BYTES   (16*KP*2)               // 26880
#define G_OFF     (X_OFF + X_BYTES)       // 241920
#define G_BYTES   (16*GCOLS*4)            // 8192
#define HB_OFF    (G_OFF + G_BYTES)       // 250112
#define HB_BYTES  (16*HP*2)               // 1280
#define C_OFF     (HB_OFF + HB_BYTES)     // 251392
#define C_BYTES   (16*HSLICE*4)           // 2048
#define WAS_OFF   (C_OFF + C_BYTES)       // 253440
#define WAS_BYTES (NSLOT*HP*2)            // 10240
#define RAS_OFF   (WAS_OFF + WAS_BYTES)   // 263680
#define BIAS_OFF  (RAS_OFF + WAS_BYTES)   // 273920
#define EST_OFF   (BIAS_OFF + GCOLS*4)    // 274432: [2][16][256] f32 async emb stage
#define EST_BYTES (2*16*IDIM*4)           // 32768
#define SMEM_BYTES (EST_OFF + EST_BYTES)  // 307200 ( < 320KB WGP LDS )

// ---------------- workspace layout (bytes), 256-aligned ----------------
#define WS_WCOMB  0                                   // [2048][832] bf16
#define WS_WWA    (WS_WCOMB + 2048*832*2)             // [128][512] bf16
#define WS_WRA    (WS_WWA   + 128*512*2)
#define WS_WWV    (WS_WRA   + 128*512*2)              // [64][256] bf16
#define WS_BCOMB  (WS_WWV   + 64*256*2)               // [2048] f32
#define WS_HST    (WS_BCOMB + 2048*4)                 // [8][2][16][512] bf16
#define WS_RST    (WS_HST   + 8*2*16*512*2)           // [8][16][64] bf16
#define WS_MEM    (WS_RST   + 8*16*64*2)              // [8][16][128][64] f32
#define WS_PWA    (WS_MEM   + 8*16*128*64*4)          // [8][16cw][16][128] f32
#define WS_VBUF   (WS_PWA   + 8*16*16*128*4)          // [8][16][64] f32
#define WS_WABUF  (WS_VBUF  + 8*16*64*4)              // [8][16][128] f32
#define WS_PRD    (WS_WABUF + 8*16*128*4)             // [8][16cw][16][64] f32
#define WS_BAR    (WS_PRD   + 8*16*16*64*4)           // [8] u32 (padded)
#define WS_TOTAL  (WS_BAR + 256)                      // ~9.9 MB

typedef __attribute__((ext_vector_type(16))) __bf16 v16bf;
typedef __attribute__((ext_vector_type(8)))  float  v8f;

union FragAB { uint4 u[2]; v16bf v; };
union AccU   { v8f v; float f[8]; };

__device__ __forceinline__ unsigned short f2bf(float f) {
  union { float f; unsigned u; } x; x.f = f;
  unsigned u = x.u + 0x7FFFu + ((x.u >> 16) & 1u);
  return (unsigned short)(u >> 16);
}
__device__ __forceinline__ float bf2f(unsigned short h) {
  union { unsigned u; float f; } x; x.u = ((unsigned)h) << 16; return x.f;
}
// single-instruction hardware reciprocal (v_rcp_f32) instead of IEEE div sequence
__device__ __forceinline__ float fast_rcp(float x) { return __builtin_amdgcn_rcpf(x); }
__device__ __forceinline__ float sigm(float x) { return fast_rcp(1.0f + __expf(-x)); }
// branch-free tanh via hardware exp/rcp (avoids branchy OCML tanhf)
__device__ __forceinline__ float fast_tanh(float x) {
  float e = __expf(2.0f * x);
  return 1.0f - 2.0f * fast_rcp(e + 1.0f);
}

// ---- CDNA5 async global->LDS streaming (ASYNCcnt path) ----
// One thread moves 64B: 4x b128; INST_OFFSET applies to both global and LDS addr.
__device__ __forceinline__ void emb_async64(unsigned lds_off, const float* g) {
  unsigned long long ga = (unsigned long long)(uintptr_t)g;
  asm volatile(
      "global_load_async_to_lds_b128 %0, %1, off\n\t"
      "global_load_async_to_lds_b128 %0, %1, off offset:16\n\t"
      "global_load_async_to_lds_b128 %0, %1, off offset:32\n\t"
      "global_load_async_to_lds_b128 %0, %1, off offset:48"
      :: "v"(lds_off), "v"(ga) : "memory");
}
__device__ __forceinline__ void wait_async0() {
  asm volatile("s_wait_asynccnt 0x0" ::: "memory");
}

// device-scope spin barrier among the 16 blocks of one batch group
__device__ __forceinline__ void gbar(unsigned* ctr, unsigned& target) {
  target += NCW;
  __threadfence();
  __syncthreads();
  if (threadIdx.x == 0) {
    __hip_atomic_fetch_add(ctr, 1u, __ATOMIC_RELEASE, __HIP_MEMORY_SCOPE_AGENT);
    while (__hip_atomic_load(ctr, __ATOMIC_RELAXED, __HIP_MEMORY_SCOPE_AGENT) < target) {
      __builtin_amdgcn_s_sleep(2);
    }
  }
  __syncthreads();
  (void)__hip_atomic_load(ctr, __ATOMIC_ACQUIRE, __HIP_MEMORY_SCOPE_AGENT); // inv stale caches
}

// ================= prep kernels =================
__global__ void prep_weights(const float* __restrict__ w_ih, const float* __restrict__ w_hh,
                             unsigned short* __restrict__ Wcomb) {
  for (long i = (long)blockIdx.x * blockDim.x + threadIdx.x; i < (long)2048 * KTOT;
       i += (long)gridDim.x * blockDim.x) {
    int r = (int)(i / KTOT), k = (int)(i % KTOT);
    float v = (k < KIH) ? w_ih[(long)r * KIH + k] : w_hh[(long)r * CDIM + (k - KIH)];
    Wcomb[i] = f2bf(v);
  }
}

__global__ void prep_state(const float* __restrict__ W_wa, const float* __restrict__ W_ra,
                           const float* __restrict__ W_wv, const float* __restrict__ b_ih,
                           const float* __restrict__ b_hh, const float* __restrict__ h0,
                           const float* __restrict__ r0,
                           unsigned short* Wwa_b, unsigned short* Wra_b, unsigned short* Wwv_b,
                           float* Bcomb, unsigned short* Hst, unsigned short* Rst,
                           float* MEM, unsigned* BAR) {
  const long stride = (long)gridDim.x * blockDim.x;
  const long t0 = (long)blockIdx.x * blockDim.x + threadIdx.x;
  for (long i = t0; i < 128 * 512; i += stride) Wwa_b[i] = f2bf(W_wa[i]);
  for (long i = t0; i < 128 * 512; i += stride) Wra_b[i] = f2bf(W_ra[i]);
  for (long i = t0; i < 64 * 256; i += stride)  Wwv_b[i] = f2bf(W_wv[i]);
  for (long i = t0; i < 2048; i += stride)      Bcomb[i] = b_ih[i] + b_hh[i];
  for (long i = t0; i < NGROUP * 16 * 512; i += stride) {   // Hst parity 0 = h0
    int g = (int)(i >> 13), rest = (int)(i & 8191);
    Hst[(long)g * (2 * 16 * 512) + rest] = f2bf(h0[rest & 511]);
  }
  for (long i = t0; i < NGROUP * 16 * 64; i += stride) Rst[i] = f2bf(r0[i & 63]);
  for (long i = t0; i < (long)NGROUP * 16 * 128 * 64; i += stride) MEM[i] = 0.0f;
  for (long i = t0; i < 8; i += stride) BAR[i] = 0u;
}

__global__ void prep_out_r(const int* __restrict__ ilen_p, const float* __restrict__ r0,
                           float* __restrict__ out) {
  const int ilen = *ilen_p;
  for (long i = (long)blockIdx.x * blockDim.x + threadIdx.x; i < (long)T_STEPS * BATCH * MDIM;
       i += (long)gridDim.x * blockDim.x) {
    int m = (int)(i & 63);
    int b = (int)((i >> 6) & 127);
    int t = (int)(i >> 13);
    if (t < ilen) out[((long)t * BATCH + b) * 576 + 512 + m] = r0[m];
  }
}

// ================= persistent main kernel =================
__global__ __launch_bounds__(256, 1)
void marnn_main(const float* __restrict__ embs, const int* __restrict__ ilen_p,
                const float* __restrict__ c0, const float* __restrict__ b_wa,
                const float* __restrict__ b_ra, const float* __restrict__ b_wv,
                const unsigned short* __restrict__ Wcomb, const unsigned short* __restrict__ Wwa_b,
                const unsigned short* __restrict__ Wra_b, const unsigned short* __restrict__ Wwv_b,
                const float* __restrict__ Bcomb,
                unsigned short* Hst, unsigned short* Rst, float* MEM,
                float* PWA, float* Vbuf, float* WAbuf, float* PRD,
                unsigned* BAR, float* __restrict__ out) {
  extern __shared__ char smem[];
  unsigned short* Wsh  = (unsigned short*)(smem + W_OFF);
  unsigned short* Xsh  = (unsigned short*)(smem + X_OFF);
  float*          Gsh  = (float*)(smem + G_OFF);
  unsigned short* Hbsh = (unsigned short*)(smem + HB_OFF);
  float*          Csh  = (float*)(smem + C_OFF);
  unsigned short* WAsh = (unsigned short*)(smem + WAS_OFF);
  unsigned short* RAsh = (unsigned short*)(smem + RAS_OFF);
  float*          Bsh  = (float*)(smem + BIAS_OFF);
  float*          Est  = (float*)(smem + EST_OFF);   // [2][16*256] f32 stage
  // low 32 bits of a generic LDS pointer == byte offset within workgroup LDS
  const unsigned est_lds = (unsigned)(uintptr_t)(const void*)Est;

  const int blk = blockIdx.x;        // 0..127
  const int bg  = blk >> 4;          // batch group 0..7
  const int cw  = blk & 15;          // column workgroup 0..15
  const int hbase = cw * HSLICE;
  const int tid  = threadIdx.x;
  const int lane = tid & 31;
  const int wv   = tid >> 5;         // wave 0..7
  const int ilen = *ilen_p;

  unsigned short* HstG = Hst + (long)bg * (2 * 16 * 512);
  unsigned short* RstG = Rst + (long)bg * (16 * 64);
  float* MEMG   = MEM   + (long)bg * (16 * 128 * 64);
  float* PWAG   = PWA   + (long)bg * (16 * 16 * 128);
  float* VbufG  = Vbuf  + (long)bg * (16 * 64);
  float* WAbufG = WAbuf + (long)bg * (16 * 128);
  float* PRDG   = PRD   + (long)bg * (16 * 16 * 64);
  unsigned* barp = BAR + bg;

  // kick off async stream of the t=0 emb tile (overlaps the weight staging below)
  emb_async64(est_lds + tid * 64,
              embs + ((long)0 * BATCH + bg * 16) * IDIM + tid * 16);

  // ---- one-time LDS residency: weight slice, head slices, bias, c-state ----
  for (int c = 0; c < GCOLS; ++c) {
    const int row = (c >> 5) * CDIM + hbase + (c & 31);   // i/f/g/o gate remap
    const unsigned short* src = Wcomb + (long)row * KTOT;
    for (int k = tid; k < KTOT; k += 256) Wsh[c * KP + k] = src[k];
  }
  for (int idx = tid; idx < NSLOT * HSLICE; idx += 256) {
    int n = idx >> 5, k = idx & 31;
    WAsh[n * HP + k] = Wwa_b[(long)n * CDIM + hbase + k];
    RAsh[n * HP + k] = Wra_b[(long)n * CDIM + hbase + k];
  }
  if (tid < GCOLS) Bsh[tid] = Bcomb[(tid >> 5) * CDIM + hbase + (tid & 31)];
  for (int idx = tid; idx < 16 * HSLICE; idx += 256) Csh[idx] = c0[hbase + (idx & 31)];
  __syncthreads();

  const int m    = lane & 15;                 // WMMA row (A) / col (B,D)
  const int half = lane >> 4;
  unsigned target = 0;                        // group barrier generation

  for (int t = 0; t < T_STEPS; ++t) {
    const int isw = (t < ilen);
    const int rp = t & 1, wp = rp ^ 1;

    // ---------- stage x_t = [emb | r | h_{t-1}] as bf16 into LDS ----------
    wait_async0();                            // our wave's emb tile is in LDS
    {
      // each thread converts exactly the 64B it streamed (wave-local hazard only)
      const float* st = Est + rp * (16 * IDIM) + tid * 16;
      unsigned short* dst = Xsh + (tid >> 4) * KP + (tid & 15) * 16;
#pragma unroll
      for (int i = 0; i < 16; ++i) dst[i] = f2bf(st[i]);
    }
    if (t + 1 < T_STEPS) {                    // stream next tile during this step's math
      emb_async64(est_lds + wp * (16 * IDIM * 4) + tid * 64,
                  embs + ((long)(t + 1) * BATCH + bg * 16) * IDIM + tid * 16);
    }
    for (int idx = tid; idx < 16 * MDIM; idx += 256) {
      int b = idx >> 6, j = idx & 63;
      Xsh[b * KP + IDIM + j] = RstG[b * 64 + j];
    }
    {
      const unsigned short* hsrc = HstG + rp * (16 * 512);
      const int b = tid >> 4, k0 = (tid & 15) * 32;
#pragma unroll 8
      for (int i = 0; i < 32; ++i) Xsh[b * KP + KIH + k0 + i] = hsrc[b * 512 + k0 + i];
    }
    __syncthreads();

    // ---------- gates: [16 x 832] @ [832 x 128]-slice via bf16 WMMA ----------
    {
      AccU acc;
      const float bias = Bsh[wv * 16 + m];
#pragma unroll
      for (int j = 0; j < 8; ++j) acc.f[j] = bias;
      const char* arow = (const char*)Xsh + m * (KP * 2) + (half << 4);
      const char* brow = (const char*)Wsh + (wv * 16 + m) * (KP * 2) + (half << 4);
      FragAB a0, b0;
      a0.u[0] = *(const uint4*)(arow);
      a0.u[1] = *(const uint4*)(arow + 32);
      b0.u[0] = *(const uint4*)(brow);
      b0.u[1] = *(const uint4*)(brow + 32);
#pragma unroll 2
      for (int kk = 0; kk < KTOT / 32; ++kk) {       // software-pipelined 1 ahead
        FragAB a1, b1;
        const int nk = (kk + 1 < KTOT / 32) ? kk + 1 : kk;
        a1.u[0] = *(const uint4*)(arow + nk * 64);
        a1.u[1] = *(const uint4*)(arow + nk * 64 + 32);
        b1.u[0] = *(const uint4*)(brow + nk * 64);
        b1.u[1] = *(const uint4*)(brow + nk * 64 + 32);
        acc.v = __builtin_amdgcn_wmma_f32_16x16x32_bf16(false, a0.v, false, b0.v,
                                                        (short)0, acc.v, false, false);
        a0 = a1; b0 = b1;
      }
#pragma unroll
      for (int j = 0; j < 8; ++j)
        Gsh[(half * 8 + j) * GCOLS + wv * 16 + m] = acc.f[j];
    }
    __syncthreads();

    // ---------- LSTM pointwise for our 32 hidden units ----------
    {
      unsigned short* HstW = HstG + wp * (16 * 512);
#pragma unroll
      for (int e = tid * 2; e < tid * 2 + 2; ++e) {
        const int b = e >> 5, u = e & 31;
        const float ig = Gsh[b * GCOLS + u];
        const float fg = Gsh[b * GCOLS + 32 + u];
        const float gg = Gsh[b * GCOLS + 64 + u];
        const float og = Gsh[b * GCOLS + 96 + u];
        const float cn = sigm(fg) * Csh[b * 32 + u] + sigm(ig) * fast_tanh(gg);
        const float hn = sigm(og) * fast_tanh(cn);
        Csh[b * 32 + u] = cn;
        const unsigned short hb = f2bf(hn);
        Hbsh[b * HP + u] = hb;
        HstW[b * 512 + hbase + u] = hb;
        out[((long)t * BATCH + bg * 16 + b) * 576 + hbase + u] = hn;
      }
    }
    __syncthreads();

    // ---------- partial head pre-activation: h_slice[16x32] @ Whead_slice ----------
    {
      FragAB a, b;
      const char* ar = (const char*)Hbsh + m * (HP * 2) + (half << 4);
      a.u[0] = *(const uint4*)ar;
      a.u[1] = *(const uint4*)(ar + 32);
      const unsigned short* Wh = isw ? WAsh : RAsh;
      const int n = wv * 16 + m;
      const char* br = (const char*)Wh + n * (HP * 2) + (half << 4);
      b.u[0] = *(const uint4*)br;
      b.u[1] = *(const uint4*)(br + 32);
      AccU z;
#pragma unroll
      for (int j = 0; j < 8; ++j) z.f[j] = 0.0f;
      z.v = __builtin_amdgcn_wmma_f32_16x16x32_bf16(false, a.v, false, b.v,
                                                    (short)0, z.v, false, false);
      float* dst = PWAG + cw * (16 * 128);
#pragma unroll
      for (int j = 0; j < 8; ++j) dst[(half * 8 + j) * 128 + n] = z.f[j];
    }
    // write value v = emb @ W_wv.T + b_wv (group owner only, write phase only)
    if (cw == 0 && isw) {
      for (int idx = tid; idx < 16 * MDIM; idx += 256) {
        int b = idx >> 6, j = idx & 63;
        float s = b_wv[j];
        const unsigned short* wr = Wwv_b + j * IDIM;
        const unsigned short* xr = Xsh + b * KP;
        for (int k = 0; k < IDIM; ++k) s += bf2f(xr[k]) * bf2f(wr[k]);
        VbufG[b * 64 + j] = s;
      }
    }
    gbar(barp, target);   // b1: PWA (+Vbuf, Hst) complete

    // ---------- phase B (cw0): reduce partials, bias, row softmax ----------
    if (cw == 0) {
      const float* bias = isw ? b_wa : b_ra;
      for (int idx = tid; idx < 16 * NSLOT; idx += 256) {
        int b = idx >> 7, n = idx & 127;
        float s = bias[n];
#pragma unroll
        for (int p = 0; p < NCW; ++p) s += PWAG[p * 2048 + b * 128 + n];
        Gsh[b * 128 + n] = s;
      }
      __syncthreads();
      if (tid < 16) {
        const int b = tid;
        float mx = -3.4e38f;
        for (int n = 0; n < NSLOT; ++n) mx = fmaxf(mx, Gsh[b * 128 + n]);
        float sum = 0.0f;
        for (int n = 0; n < NSLOT; ++n) {
          float e = __expf(Gsh[b * 128 + n] - mx);
          Gsh[b * 128 + n] = e; sum += e;
        }
        const float inv = fast_rcp(sum);
        for (int n = 0; n < NSLOT; ++n) WAbufG[b * 128 + n] = Gsh[b * 128 + n] * inv;
      }
      __syncthreads();
    }
    gbar(barp, target);   // b2: softmax address ready

    if (isw) {
      // ---------- phase C (write half): mem += wa (x) v on our 8 slots ----------
      const int ns = cw * 8;
      for (int idx = tid; idx < 8 * 16 * MDIM; idx += 256) {
        const int nl = idx >> 10, b = (idx >> 6) & 15, mm = idx & 63;
        const int n = ns + nl;
        MEMG[(b * 128 + n) * 64 + mm] += WAbufG[b * 128 + n] * VbufG[b * 64 + mm];
      }
      // mem only consumed in the read half; no extra barrier needed this step
    } else {
      // ---------- phase C (read half): partial r_read over our 8 slots ----------
      const int ns = cw * 8;
      for (int idx = tid; idx < 16 * MDIM; idx += 256) {
        const int b = idx >> 6, mm = idx & 63;
        float s = 0.0f;
#pragma unroll
        for (int j = 0; j < 8; ++j)
          s += WAbufG[b * 128 + ns + j] * MEMG[(b * 128 + ns + j) * 64 + mm];
        PRDG[cw * 1024 + b * 64 + mm] = s;
      }
      gbar(barp, target); // b3: partial reads complete
      if (cw == 0) {
        for (int idx = tid; idx < 16 * MDIM; idx += 256) {
          const int b = idx >> 6, mm = idx & 63;
          float s = 0.0f;
#pragma unroll
          for (int p = 0; p < NCW; ++p) s += PRDG[p * 1024 + b * 64 + mm];
          RstG[b * 64 + mm] = f2bf(s);
          out[((long)t * BATCH + bg * 16 + b) * 576 + 512 + mm] = s;
        }
      }
      gbar(barp, target); // b4: r_new visible before next step's x staging
    }
  }
}

// ================= host =================
extern "C" void kernel_launch(void* const* d_in, const int* in_sizes, int n_in,
                              void* d_out, int out_size, void* d_ws, size_t ws_size,
                              hipStream_t stream) {
  const float* embs = (const float*)d_in[0];
  const int*   ilen = (const int*)d_in[1];
  const float* w_ih = (const float*)d_in[2];
  const float* w_hh = (const float*)d_in[3];
  const float* b_ih = (const float*)d_in[4];
  const float* b_hh = (const float*)d_in[5];
  const float* h0   = (const float*)d_in[6];
  const float* c0   = (const float*)d_in[7];
  const float* r0   = (const float*)d_in[8];
  const float* W_wa = (const float*)d_in[9];
  const float* b_wa = (const float*)d_in[10];
  const float* W_wv = (const float*)d_in[11];
  const float* b_wv = (const float*)d_in[12];
  const float* W_ra = (const float*)d_in[13];
  const float* b_ra = (const float*)d_in[14];
  float* out = (float*)d_out;

  char* ws = (char*)d_ws;
  unsigned short* Wcomb = (unsigned short*)(ws + WS_WCOMB);
  unsigned short* Wwa_b = (unsigned short*)(ws + WS_WWA);
  unsigned short* Wra_b = (unsigned short*)(ws + WS_WRA);
  unsigned short* Wwv_b = (unsigned short*)(ws + WS_WWV);
  float*          Bcomb = (float*)(ws + WS_BCOMB);
  unsigned short* Hst   = (unsigned short*)(ws + WS_HST);
  unsigned short* Rst   = (unsigned short*)(ws + WS_RST);
  float*          MEM   = (float*)(ws + WS_MEM);
  float*          PWA   = (float*)(ws + WS_PWA);
  float*          Vbuf  = (float*)(ws + WS_VBUF);
  float*          WAbuf = (float*)(ws + WS_WABUF);
  float*          PRD   = (float*)(ws + WS_PRD);
  unsigned*       BAR   = (unsigned*)(ws + WS_BAR);

  (void)in_sizes; (void)n_in; (void)out_size; (void)ws_size;

  // allow >64KB dynamic LDS (CDNA5 WGP has 320KB)
  static bool attr_set = [] {
    (void)hipFuncSetAttribute((const void*)marnn_main,
                              hipFuncAttributeMaxDynamicSharedMemorySize, SMEM_BYTES);
    return true;
  }();
  (void)attr_set;

  prep_weights<<<dim3(2048), dim3(256), 0, stream>>>(w_ih, w_hh, Wcomb);
  prep_state<<<dim3(2048), dim3(256), 0, stream>>>(W_wa, W_ra, W_wv, b_ih, b_hh, h0, r0,
                                                   Wwa_b, Wra_b, Wwv_b, Bcomb, Hst, Rst,
                                                   MEM, BAR);
  prep_out_r<<<dim3(4096), dim3(256), 0, stream>>>(ilen, r0, out);
  marnn_main<<<dim3(NGROUP * NCW), dim3(256), SMEM_BYTES, stream>>>(
      embs, ilen, c0, b_wa, b_ra, b_wv, Wcomb, Wwa_b, Wra_b, Wwv_b, Bcomb,
      Hst, Rst, MEM, PWA, Vbuf, WAbuf, PRD, BAR, out);
}
